// MultiRelGraphConv_83880711291264
// MI455X (gfx1250) — compile-verified, matrix operand order checked by reading, not run on hardware
//
#include <hip/hip_runtime.h>

typedef __attribute__((ext_vector_type(16))) _Float16 v16h;
typedef __attribute__((ext_vector_type(8)))  _Float16 v8h;
typedef __attribute__((ext_vector_type(8)))  float    v8f;

constexpr int N_NODES = 30000;
constexpr int N_EDGES = 300000;
constexpr int H   = 128;
constexpr int H3  = 384;
constexpr int R   = 200;
constexpr int L   = 2;
constexpr float RRELU_SLOPE = (1.0f / 8.0f + 1.0f / 3.0f) * 0.5f;

// ---------------------------------------------------------------------------
// WMMA helpers (CDNA5 wave32, V_WMMA_F32_16X16X32_F16)
// 16-bit A 16x32: lane l holds row m=l&15 (half=l>>4); per lane the 16 halfs
// are two contiguous runs of 8 at k0 = 8*half and k0+16. B 32x16 symmetric
// with n = l&15.
// ---------------------------------------------------------------------------
__device__ __forceinline__ v16h load_frag(const _Float16* rowBase, int lane) {
  const int half = (lane >> 4) & 1;
  const _Float16* q = rowBase + half * 8;
  v8h lo = *(const v8h*)(q);
  v8h hi = *(const v8h*)(q + 16);
  v16h r;
#pragma unroll
  for (int i = 0; i < 8; ++i) { r[i] = lo[i]; r[8 + i] = hi[i]; }
  return r;
}

__device__ __forceinline__ v8f wmma_f16(v16h a, v16h b, v8f c) {
  return __builtin_amdgcn_wmma_f32_16x16x32_f16(false, a, false, b, (short)0, c,
                                                false, false);
}

// CDNA5 async global->LDS copy (16 bytes per active lane), ASYNCcnt-tracked.
__device__ __forceinline__ void async_copy16(void* ldst, const void* gsrc) {
  asm volatile("global_load_async_to_lds_b128 %0, %1, off"
               :
               : "v"((unsigned)(size_t)ldst), "v"(gsrc)
               : "memory");
}
__device__ __forceinline__ void async_wait0() {
  asm volatile("s_wait_asynccnt 0x0" ::: "memory");
}

// ---------------------------------------------------------------------------
// Utility kernels
// ---------------------------------------------------------------------------
__global__ void k_f32_to_f16(const float* __restrict__ in,
                             _Float16* __restrict__ out, int n) {
  int i = blockIdx.x * 256 + threadIdx.x;
  if (i < n) out[i] = (_Float16)in[i];
}

__global__ void k_mask_scatter(const int* __restrict__ src,
                               const int* __restrict__ dst,
                               const int* __restrict__ ety,
                               unsigned* __restrict__ maskbits) {
  int e = blockIdx.x * 256 + threadIdx.x;
  if (e >= N_EDGES) return;
  unsigned base = (unsigned)ety[e] * (unsigned)N_NODES;
  unsigned p0 = base + (unsigned)src[e];
  unsigned p1 = base + (unsigned)dst[e];
  atomicOr(&maskbits[p0 >> 5], 1u << (p0 & 31));
  atomicOr(&maskbits[p1 >> 5], 1u << (p1 & 31));
}

__global__ void k_rel_reduce(const float* __restrict__ nf,
                             const unsigned* __restrict__ maskbits,
                             float* __restrict__ rel_sum,
                             float* __restrict__ rel_cnt) {
  const int r = blockIdx.x;            // relation
  const int chunk = blockIdx.y;        // node-range chunk (8 chunks)
  const int h = threadIdx.x;           // 128 threads = one column each
  const int per = (N_NODES + 7) / 8;
  int n0 = chunk * per;
  int n1 = n0 + per; if (n1 > N_NODES) n1 = N_NODES;
  const unsigned rowbase = (unsigned)r * (unsigned)N_NODES;
  float acc = 0.0f, cnt = 0.0f;
  for (int n = n0; n < n1; ++n) {
    unsigned p = rowbase + (unsigned)n;
    if (maskbits[p >> 5] & (1u << (p & 31))) {
      acc += nf[(size_t)n * H + h];
      cnt += 1.0f;
    }
  }
  atomicAdd(&rel_sum[r * H + h], acc);
  if (h == 0) atomicAdd(&rel_cnt[r], cnt);
}

__global__ void k_rel_finalize(const float* __restrict__ rel_sum,
                               const float* __restrict__ rel_cnt,
                               _Float16* __restrict__ re16) {
  int i = blockIdx.x * 256 + threadIdx.x;
  if (i >= R * H) return;
  float c = rel_cnt[i / H];
  float v = (c > 0.0f) ? rel_sum[i] / c : 0.0f;
  re16[i] = (_Float16)v;
}

__global__ void k_deg(const int* __restrict__ dst, float* __restrict__ deg) {
  int e = blockIdx.x * 256 + threadIdx.x;
  if (e < N_EDGES) atomicAdd(&deg[dst[e]], 1.0f);
}

// ---------------------------------------------------------------------------
// Edge GEMM: out[e,0:128] = concat(re[ety[e]], nf[src[e]], X3) @ W^T + b
// mode 0: X3 = ef[e],       output atomically accumulated into agg[dst[e],:]
// mode 1: X3 = nf[dst[e]],  output stored as f16 (e_in)
// Block: 256 threads = 8 waves, 64 edges, A staged in LDS (64x384 f16) via
// CDNA5 async global->LDS gather copies. W is f16 row-major [128 out][384 k].
// ---------------------------------------------------------------------------
__global__ __launch_bounds__(256) void k_edge_gemm(
    const _Float16* __restrict__ re16, const _Float16* __restrict__ nf16,
    const _Float16* __restrict__ ef16, const int* __restrict__ src,
    const int* __restrict__ dst, const int* __restrict__ ety,
    const _Float16* __restrict__ W, const float* __restrict__ bias,
    float* __restrict__ agg, _Float16* __restrict__ out16, int mode) {
  __shared__ _Float16 sA[64 * H3];
  __shared__ int sSrc[64], sDst[64], sEty[64];
  const int tid = threadIdx.x;
  const int eBase = blockIdx.x * 64;

  if (tid < 64) {
    int e = eBase + tid;
    sSrc[tid] = (e < N_EDGES) ? src[e] : 0;
    sDst[tid] = (e < N_EDGES) ? dst[e] : 0;
    sEty[tid] = (e < N_EDGES) ? ety[e] : 0;
  }
  __syncthreads();

  // Stage gathered A rows: 64 rows x 48 16-byte chunks, async to LDS.
  for (int idx = tid; idx < 64 * 48; idx += 256) {
    int row = idx / 48;
    int c = idx - row * 48;      // 16B chunk within the 768B row
    int e = eBase + row;
    if (e < N_EDGES) {
      int seg = c >> 4;          // which 128-f16 segment
      int off8 = (c & 15) * 8;   // f16 offset within segment
      const _Float16* gsrc;
      if (seg == 0)      gsrc = re16 + (size_t)sEty[row] * H + off8;
      else if (seg == 1) gsrc = nf16 + (size_t)sSrc[row] * H + off8;
      else               gsrc = (mode == 0)
                                    ? ef16 + (size_t)e * H + off8
                                    : nf16 + (size_t)sDst[row] * H + off8;
      async_copy16(&sA[row * H3 + c * 8], gsrc);
    }
  }
  async_wait0();
  __syncthreads();

  const int w = tid >> 5, lane = tid & 31;
  const int mtile = w >> 1;        // 4 row-tiles of 16 edges
  const int ngrp = w & 1;          // 2 groups x 4 col-tiles = 128 outputs
  const int mrow = mtile * 16 + (lane & 15);

  v8f acc[4] = {};
#pragma unroll 1
  for (int ks = 0; ks < 12; ++ks) {              // K = 384
    v16h a = load_frag(&sA[mrow * H3 + ks * 32], lane);
#pragma unroll
    for (int t = 0; t < 4; ++t) {
      int n = ngrp * 64 + t * 16 + (lane & 15);
      v16h b = load_frag(&W[(size_t)n * H3 + ks * 32], lane);  // L2-resident
      acc[t] = wmma_f16(a, b, acc[t]);
    }
  }

  const int half = lane >> 4, ncol = lane & 15;
#pragma unroll
  for (int t = 0; t < 4; ++t) {
    int n = ngrp * 64 + t * 16 + ncol;
    float bv = bias[n];
#pragma unroll
    for (int r2 = 0; r2 < 8; ++r2) {
      int m = mtile * 16 + r2 + 8 * half;
      int e = eBase + m;
      if (e < N_EDGES) {
        float v = acc[t][r2] + bv;
        if (mode == 0) atomicAdd(&agg[(size_t)sDst[m] * H + n], v);
        else           out16[(size_t)e * H + n] = (_Float16)v;
      }
    }
  }
}

// ---------------------------------------------------------------------------
// Node update: new_nf = rrelu(agg/max(deg,1) + nf @ W3^T + b3)
// ---------------------------------------------------------------------------
__global__ __launch_bounds__(256) void k_node_update(
    const _Float16* __restrict__ nf16, const _Float16* __restrict__ W3,
    const float* __restrict__ b3, const float* __restrict__ agg,
    const float* __restrict__ deg, float* __restrict__ nfOut) {
  __shared__ _Float16 sA[64 * H];
  const int tid = threadIdx.x;
  const int rBase = blockIdx.x * 64;
  // Stage 64 node rows (64 x 16 chunks of 16B) async to LDS.
  for (int idx = tid; idx < 64 * 16; idx += 256) {
    int row = idx >> 4;
    int c = idx & 15;
    int nrow = rBase + row;
    if (nrow < N_NODES)
      async_copy16(&sA[row * H + c * 8], nf16 + (size_t)nrow * H + c * 8);
  }
  async_wait0();
  __syncthreads();

  const int w = tid >> 5, lane = tid & 31;
  const int mtile = w >> 1, ngrp = w & 1;
  const int mrow = mtile * 16 + (lane & 15);

  v8f acc[4] = {};
#pragma unroll
  for (int ks = 0; ks < 4; ++ks) {               // K = 128
    v16h a = load_frag(&sA[mrow * H + ks * 32], lane);
#pragma unroll
    for (int t = 0; t < 4; ++t) {
      int n = ngrp * 64 + t * 16 + (lane & 15);
      v16h b = load_frag(&W3[(size_t)n * H + ks * 32], lane);
      acc[t] = wmma_f16(a, b, acc[t]);
    }
  }

  const int half = lane >> 4, ncol = lane & 15;
#pragma unroll
  for (int t = 0; t < 4; ++t) {
    int n = ngrp * 64 + t * 16 + ncol;
    float bv = b3[n];
#pragma unroll
    for (int r2 = 0; r2 < 8; ++r2) {
      int row = rBase + mtile * 16 + r2 + 8 * half;
      if (row < N_NODES) {
        float d = deg[row]; d = d > 1.0f ? d : 1.0f;
        float pre = acc[t][r2] + bv + agg[(size_t)row * H + n] / d;
        nfOut[(size_t)row * H + n] = pre >= 0.0f ? pre : RRELU_SLOPE * pre;
      }
    }
  }
}

// ---------------------------------------------------------------------------
// Fused GRU cell for edges: out = GRU(x, h); Wih/Whh are f16 [384][128].
// Block = 256 threads (8 waves), 16 edges; gi/gh (16x384 f32) staged in LDS.
// ---------------------------------------------------------------------------
__global__ __launch_bounds__(256) void k_gru(
    const _Float16* __restrict__ x16, const _Float16* __restrict__ h16,
    const float* __restrict__ h32, const _Float16* __restrict__ Wih,
    const _Float16* __restrict__ Whh, const float* __restrict__ bih,
    const float* __restrict__ bhh, _Float16* __restrict__ out16,
    float* __restrict__ out32) {
  __shared__ _Float16 sX[16 * H];
  __shared__ _Float16 sH[16 * H];
  __shared__ float sGi[16 * H3];
  __shared__ float sGh[16 * H3];
  const int tid = threadIdx.x;
  const size_t eBase = (size_t)blockIdx.x * 16;
  // Stage x and h tiles (2 x 256 chunks of 16B) async to LDS.
  for (int idx = tid; idx < 512; idx += 256) {
    int buf = idx >> 8;
    int c = idx & 255;
    const _Float16* g = (buf ? h16 : x16) + eBase * H + c * 8;
    _Float16* s = (buf ? sH : sX) + c * 8;
    async_copy16(s, g);
  }
  async_wait0();
  __syncthreads();

  const int w = tid >> 5, lane = tid & 31;
  const int mrow = lane & 15;
  const int half = lane >> 4, ncol = lane & 15;

  for (int nt = w * 3; nt < w * 3 + 3; ++nt) {   // 24 n-tiles over 3H
    v8f gi = {}, gh = {};
#pragma unroll
    for (int ks = 0; ks < 4; ++ks) {             // K = 128
      v16h ax = load_frag(&sX[mrow * H + ks * 32], lane);
      v16h ah = load_frag(&sH[mrow * H + ks * 32], lane);
      int n = nt * 16 + (lane & 15);
      v16h bi = load_frag(&Wih[(size_t)n * H + ks * 32], lane);
      v16h bh = load_frag(&Whh[(size_t)n * H + ks * 32], lane);
      gi = wmma_f16(ax, bi, gi);
      gh = wmma_f16(ah, bh, gh);
    }
    int n = nt * 16 + ncol;
#pragma unroll
    for (int r2 = 0; r2 < 8; ++r2) {
      int m = r2 + 8 * half;
      sGi[m * H3 + n] = gi[r2];
      sGh[m * H3 + n] = gh[r2];
    }
  }
  __syncthreads();

  for (int idx = tid; idx < 16 * H; idx += 256) {
    int m = idx >> 7, h = idx & 127;
    size_t e = eBase + m;
    float gir = sGi[m * H3 + h] + bih[h];
    float ghr = sGh[m * H3 + h] + bhh[h];
    float giz = sGi[m * H3 + 128 + h] + bih[128 + h];
    float ghz = sGh[m * H3 + 128 + h] + bhh[128 + h];
    float gin = sGi[m * H3 + 256 + h] + bih[256 + h];
    float ghn = sGh[m * H3 + 256 + h] + bhh[256 + h];
    float r = 1.0f / (1.0f + expf(-(gir + ghr)));
    float z = 1.0f / (1.0f + expf(-(giz + ghz)));
    float nn = tanhf(gin + r * ghn);
    float hv = h32[e * H + h];
    float o = (1.0f - z) * nn + z * hv;
    if (out16) out16[e * H + h] = (_Float16)o;
    if (out32) out32[e * H + h] = o;
  }
}

// ---------------------------------------------------------------------------
// Host orchestration
// ---------------------------------------------------------------------------
extern "C" void kernel_launch(void* const* d_in, const int* in_sizes, int n_in,
                              void* d_out, int out_size, void* d_ws,
                              size_t ws_size, hipStream_t stream) {
  (void)in_sizes; (void)n_in; (void)out_size; (void)ws_size;
  const float* nf_in = (const float*)d_in[0];
  const float* ef_in = (const float*)d_in[1];
  const int*   src   = (const int*)d_in[2];
  const int*   dst   = (const int*)d_in[3];
  const int*   ety   = (const int*)d_in[4];
  const float* W1 = (const float*)d_in[5];
  const float* b1 = (const float*)d_in[6];
  const float* W2 = (const float*)d_in[7];
  const float* b2 = (const float*)d_in[8];
  const float* W3 = (const float*)d_in[9];
  const float* b3 = (const float*)d_in[10];
  const float* Wih = (const float*)d_in[11];
  const float* Whh = (const float*)d_in[12];
  const float* bih = (const float*)d_in[13];
  const float* bhh = (const float*)d_in[14];

  char* ws = (char*)d_ws;
  size_t off = 0;
  auto alloc = [&](size_t bytes) -> void* {
    void* p = ws + off;
    off = (off + bytes + 255) & ~(size_t)255;
    return p;
  };

  const size_t maskWords = ((size_t)R * N_NODES + 31) / 32;
  _Float16* W1h  = (_Float16*)alloc((size_t)L * H * H3 * 2);
  _Float16* W2h  = (_Float16*)alloc((size_t)L * H * H3 * 2);
  _Float16* W3h  = (_Float16*)alloc((size_t)L * H * H * 2);
  _Float16* Wihh = (_Float16*)alloc((size_t)L * H3 * H * 2);
  _Float16* Whhh = (_Float16*)alloc((size_t)L * H3 * H * 2);
  unsigned* maskbits = (unsigned*)alloc(maskWords * 4);
  float*    rel_sum  = (float*)alloc((size_t)R * H * 4);
  float*    rel_cnt  = (float*)alloc((size_t)R * 4);
  _Float16* re16     = (_Float16*)alloc((size_t)R * H * 2);
  _Float16* nf16     = (_Float16*)alloc((size_t)N_NODES * H * 2);
  _Float16* ef16     = (_Float16*)alloc((size_t)N_EDGES * H * 2);
  _Float16* ein16    = (_Float16*)alloc((size_t)N_EDGES * H * 2);
  _Float16* emsg16   = (_Float16*)alloc((size_t)N_EDGES * H * 2);
  float*    agg      = (float*)alloc((size_t)N_NODES * H * 4);
  float*    deg      = (float*)alloc((size_t)N_NODES * 4);
  float*    nfA      = (float*)alloc((size_t)N_NODES * H * 4);
  float*    efA      = (float*)alloc((size_t)N_EDGES * H * 4);

  auto cvt = [&](const float* s, _Float16* d, size_t n) {
    k_f32_to_f16<<<(int)((n + 255) / 256), 256, 0, stream>>>(s, d, (int)n);
  };

  // One-time (per launch) weight conversions, all layers stacked.
  cvt(W1,  W1h,  (size_t)L * H * H3);
  cvt(W2,  W2h,  (size_t)L * H * H3);
  cvt(W3,  W3h,  (size_t)L * H * H);
  cvt(Wih, Wihh, (size_t)L * H3 * H);
  cvt(Whh, Whhh, (size_t)L * H3 * H);

  float* out_nf = (float*)d_out;
  float* out_ef = (float*)d_out + (size_t)N_NODES * H;

  const int egBlocks  = (N_EDGES + 63) / 64;
  const int ndBlocks  = (N_NODES + 63) / 64;
  const int gruBlocks = N_EDGES / 16;   // E % 16 == 0
  const int eThBlocks = (N_EDGES + 255) / 256;

  for (int l = 0; l < L; ++l) {
    const float* nf_src = (l == 0) ? nf_in : nfA;
    const float* ef_src = (l == 0) ? ef_in : efA;
    float* nf_dst = (l == L - 1) ? out_nf : nfA;
    float* ef_dst = (l == L - 1) ? out_ef : efA;

    hipMemsetAsync(maskbits, 0, maskWords * 4, stream);
    hipMemsetAsync(rel_sum, 0, (size_t)R * H * 4, stream);
    hipMemsetAsync(rel_cnt, 0, (size_t)R * 4, stream);
    hipMemsetAsync(agg, 0, (size_t)N_NODES * H * 4, stream);
    hipMemsetAsync(deg, 0, (size_t)N_NODES * 4, stream);

    cvt(nf_src, nf16, (size_t)N_NODES * H);
    cvt(ef_src, ef16, (size_t)N_EDGES * H);

    k_mask_scatter<<<eThBlocks, 256, 0, stream>>>(src, dst, ety, maskbits);
    k_rel_reduce<<<dim3(R, 8), 128, 0, stream>>>(nf_src, maskbits, rel_sum,
                                                 rel_cnt);
    k_rel_finalize<<<(R * H + 255) / 256, 256, 0, stream>>>(rel_sum, rel_cnt,
                                                            re16);
    k_deg<<<eThBlocks, 256, 0, stream>>>(dst, deg);

    // msg GEMM, fused segment-sum into agg
    k_edge_gemm<<<egBlocks, 256, 0, stream>>>(
        re16, nf16, ef16, src, dst, ety, W1h + (size_t)l * H * H3,
        b1 + (size_t)l * H, agg, ein16 /*unused*/, 0);
    // e_in GEMM -> f16
    k_edge_gemm<<<egBlocks, 256, 0, stream>>>(
        re16, nf16, ef16, src, dst, ety, W2h + (size_t)l * H * H3,
        b2 + (size_t)l * H, agg /*unused*/, ein16, 1);

    // node update (mean agg + self msg + rrelu)
    k_node_update<<<ndBlocks, 256, 0, stream>>>(
        nf16, W3h + (size_t)l * H * H, b3 + (size_t)l * H, agg, deg, nf_dst);

    // e_msg = GRU(e_in, ef)
    k_gru<<<gruBlocks, 256, 0, stream>>>(
        ein16, ef16, ef_src, Wihh + (size_t)l * H3 * H,
        Whhh + (size_t)l * H3 * H, bih + (size_t)l * H3, bhh + (size_t)l * H3,
        emsg16, nullptr);
    // new_ef = GRU(e_msg, ef)
    k_gru<<<gruBlocks, 256, 0, stream>>>(
        emsg16, ef16, ef_src, Wihh + (size_t)l * H3 * H,
        Whhh + (size_t)l * H3 * H, bih + (size_t)l * H3, bhh + (size_t)l * H3,
        nullptr, ef_dst);
  }
}
